// Attention_77214922047844
// MI455X (gfx1250) — compile-verified, matrix operand order checked by reading, not run on hardware
//
#include <hip/hip_runtime.h>
#include <hip/hip_bf16.h>

// ---------------------------------------------------------------------------
// PVT spatial-reduction attention for MI455X (gfx1250, wave32, WMMA).
// All five matmuls use v_wmma_f32_16x16x32_bf16 (f32 accumulate).
// B=4 NUM=2 N=4096 C=256 HEADS=8 HD=32 SR=4 -> n_kv=256.
// ---------------------------------------------------------------------------

typedef __bf16 bf16;
typedef __attribute__((ext_vector_type(8)))  __bf16 v8bf;
typedef __attribute__((ext_vector_type(16))) __bf16 v16bf;
typedef __attribute__((ext_vector_type(8)))  float  v8f;

union FragU { v16bf v; v8bf h[2]; };

// A/B^T fragment loader (16-bit WMMA operand, wave32).
// Per ISA 7.12.2: lane L holds row M=L%16; K-runs [b0,b0+8) and [b0+16,b0+24)
// with b0 = (L<16 ? 0 : 8)  ->  two contiguous 16-byte loads per lane.
__device__ __forceinline__ v16bf load_frag(const bf16* __restrict__ base, int ld, int lane) {
  const int m  = lane & 15;
  const int b0 = (lane & 16) ? 8 : 0;
  const bf16* p = base + m * ld + b0;
  FragU u;
  u.h[0] = *reinterpret_cast<const v8bf*>(p);
  u.h[1] = *reinterpret_cast<const v8bf*>(p + 16);
  return u.v;
}

__device__ __forceinline__ v8f wmma_bf16(v16bf a, v16bf b, v8f c) {
  return __builtin_amdgcn_wmma_f32_16x16x32_bf16(false, a, false, b, (short)0, c, false, false);
}

// ---------------------------------------------------------------------------
// f32 -> bf16 converter (optionally folds a scalar multiplier, used to fold
// the attention scale hd^-0.5 into q_w).
// ---------------------------------------------------------------------------
__global__ void __launch_bounds__(256) cvt_bf16(const float* __restrict__ src,
                                                bf16* __restrict__ dst,
                                                int n, float mult) {
  int i = blockIdx.x * blockDim.x + threadIdx.x;
  const int stride = gridDim.x * blockDim.x;
  for (; i < n; i += stride) dst[i] = (bf16)(src[i] * mult);
}

// ---------------------------------------------------------------------------
// Depthwise 4x4 stride-4 conv + bias + LayerNorm(C) -> xr_bf16 (2048 x 256).
// One 256-thread block per (bm, output-pixel); thread = channel.
// ---------------------------------------------------------------------------
__global__ void __launch_bounds__(256) conv_ln_kernel(
    const float* __restrict__ x, const float* __restrict__ sr_w,
    const float* __restrict__ sr_b, const float* __restrict__ ln_g,
    const float* __restrict__ ln_b, bf16* __restrict__ xr) {
  const int c  = threadIdx.x;
  const int bp = blockIdx.x;          // bm*256 + p
  const int bm = bp >> 8;
  const int p  = bp & 255;
  const int py = p >> 4, px = p & 15; // 16x16 reduced grid

  const float* xb = x + (size_t)bm * 4096 * 256;
  float acc = 0.f;
#pragma unroll
  for (int i = 0; i < 4; ++i)
#pragma unroll
    for (int j = 0; j < 4; ++j)
      acc += xb[((py * 4 + i) * 64 + (px * 4 + j)) * 256 + c] * sr_w[c * 16 + i * 4 + j];
  acc += sr_b[c];

  __shared__ float s1[256];
  __shared__ float s2[256];
  s1[c] = acc; s2[c] = acc * acc;
  __syncthreads();
  for (int off = 128; off > 0; off >>= 1) {
    if (c < off) { s1[c] += s1[c + off]; s2[c] += s2[c + off]; }
    __syncthreads();
  }
  const float mu  = s1[0] * (1.f / 256.f);
  const float var = s2[0] * (1.f / 256.f) - mu * mu;
  const float y = (acc - mu) * rsqrtf(var + 1e-5f) * ln_g[c] + ln_b[c];
  xr[bp * 256 + c] = (bf16)y;
}

// ---------------------------------------------------------------------------
// Generic WMMA GEMM: C(Mx N) = A(M x 256) @ W^T, W stored (N,256) row-major
// (torch Linear layout == B^T row-major, so B fragments load like A).
// Block = 128 threads = 4 waves; each wave owns a 16x64 output tile.
// MODE 0: store bf16 (q projection, scale pre-folded into W)
// MODE 1: kv projection, scatter into K (n_kv,32) and V^T (32,n_kv) per head
// MODE 2: out projection, + bias, store f32 to d_out
// ---------------------------------------------------------------------------
template <int MODE>
__global__ void __launch_bounds__(128) gemm_bf16_kernel(
    const bf16* __restrict__ A, const bf16* __restrict__ W,
    bf16* __restrict__ outb, bf16* __restrict__ outK, bf16* __restrict__ outVT,
    const float* __restrict__ bias, float* __restrict__ outf) {
  const int lane = threadIdx.x & 31;
  const int wave = threadIdx.x >> 5;
  const int row0 = blockIdx.x * 64 + wave * 16;
  const int col0 = blockIdx.y * 64;

  v8f acc[4];
#pragma unroll
  for (int t = 0; t < 4; ++t)
#pragma unroll
    for (int i = 0; i < 8; ++i) acc[t][i] = 0.f;

  const bf16* Abase = A + (size_t)row0 * 256;
#pragma unroll
  for (int kk = 0; kk < 8; ++kk) {
    const v16bf a = load_frag(Abase + kk * 32, 256, lane);
#pragma unroll
    for (int t = 0; t < 4; ++t) {
      const v16bf b = load_frag(W + (size_t)(col0 + t * 16) * 256 + kk * 32, 256, lane);
      acc[t] = wmma_bf16(a, b, acc[t]);
    }
  }

  // D layout: VGPR r, lane L -> row = r + 8*(L>=16), col = L%16.
  const int rl = lane & 15;
  const int rh = (lane >> 4) << 3;
#pragma unroll
  for (int t = 0; t < 4; ++t) {
#pragma unroll
    for (int r = 0; r < 8; ++r) {
      const int row = row0 + r + rh;
      const int col = col0 + t * 16 + rl;
      const float v = acc[t][r];
      if (MODE == 0) {
        outb[(size_t)row * 256 + col] = (bf16)v;
      } else if (MODE == 1) {
        const int s = col >> 8, h = (col >> 5) & 7, d = col & 31;
        const int p = row & 255, bm = row >> 8;
        if (s == 0) outK[((size_t)(bm * 8 + h) * 256 + p) * 32 + d] = (bf16)v;
        else        outVT[((size_t)(bm * 8 + h) * 32 + d) * 256 + p] = (bf16)v;
      } else {
        outf[(size_t)row * 256 + col] = v + bias[col];
      }
    }
  }
}

// ---------------------------------------------------------------------------
// Attention: one wave per (bm*8+h, 16-row q tile).
// QK^T: 16 wmma -> full 16x256 score block in registers.
// Softmax: register-resident, 16-lane shuffle row-reductions (D-fragment rows
// live inside one 16-lane half; xor masks 1/2/4/8 never cross halves).
// PV: P round-trips LDS (bf16) to become A fragments; 16 wmma.
// Epilogue reproduces the reference's swapaxes(1,2).reshape scramble.
// ---------------------------------------------------------------------------
__global__ void __launch_bounds__(32) attn_kernel(
    const bf16* __restrict__ q, const bf16* __restrict__ kbuf,
    const bf16* __restrict__ vtbuf, bf16* __restrict__ aout) {
  const int lane = threadIdx.x & 31;
  const int bmh = blockIdx.x;         // bm*8 + h
  const int qt  = blockIdx.y;         // 16-row q tile
  const int bm = bmh >> 3, h = bmh & 7;
  const int b = bm >> 1, m = bm & 1;

  const bf16* qbase = q + ((size_t)(bm * 4096 + qt * 16)) * 256 + h * 32;
  const bf16* kb = kbuf + (size_t)bmh * 256 * 32;   // (n_kv,32) row-major = B^T
  const bf16* vb = vtbuf + (size_t)bmh * 32 * 256;  // (32,n_kv) row-major = B^T

  const v16bf qa = load_frag(qbase, 256, lane);

  v8f z; 
#pragma unroll
  for (int i = 0; i < 8; ++i) z[i] = 0.f;

  v8f s[16];
#pragma unroll
  for (int j = 0; j < 16; ++j) {
    const v16bf kf = load_frag(kb + j * 16 * 32, 32, lane);
    s[j] = wmma_bf16(qa, kf, z);
  }

  // Row max / exp / row sum (rows spread over 16 lanes of one half-wave).
  float mx[8], sm[8];
#pragma unroll
  for (int r = 0; r < 8; ++r) mx[r] = -3.0e38f;
#pragma unroll
  for (int j = 0; j < 16; ++j)
#pragma unroll
    for (int r = 0; r < 8; ++r) mx[r] = fmaxf(mx[r], s[j][r]);
#pragma unroll
  for (int off = 8; off > 0; off >>= 1)
#pragma unroll
    for (int r = 0; r < 8; ++r) mx[r] = fmaxf(mx[r], __shfl_xor(mx[r], off, 32));
#pragma unroll
  for (int r = 0; r < 8; ++r) sm[r] = 0.f;
#pragma unroll
  for (int j = 0; j < 16; ++j)
#pragma unroll
    for (int r = 0; r < 8; ++r) {
      const float e = __expf(s[j][r] - mx[r]);
      s[j][r] = e;
      sm[r] += e;
    }
#pragma unroll
  for (int off = 8; off > 0; off >>= 1)
#pragma unroll
    for (int r = 0; r < 8; ++r) sm[r] += __shfl_xor(sm[r], off, 32);

  // P (16x256) -> LDS bf16 (padded stride), reload as A fragments.
  __shared__ __align__(16) bf16 pl[16 * 272];
  const int rl = lane & 15;
  const int rh = (lane >> 4) << 3;
#pragma unroll
  for (int j = 0; j < 16; ++j)
#pragma unroll
    for (int r = 0; r < 8; ++r)
      pl[(r + rh) * 272 + j * 16 + rl] = (bf16)s[j][r];
  __syncthreads();

  v8f o0 = z, o1 = z;
#pragma unroll
  for (int kk = 0; kk < 8; ++kk) {
    const v16bf pa = load_frag(&pl[kk * 32], 272, lane);
    const v16bf v0 = load_frag(vb + kk * 32, 256, lane);
    o0 = wmma_bf16(pa, v0, o0);
    const v16bf v1 = load_frag(vb + 16 * 256 + kk * 32, 256, lane);
    o1 = wmma_bf16(pa, v1, o1);
  }

  float inv[8];
#pragma unroll
  for (int r = 0; r < 8; ++r) inv[r] = 1.f / sm[r];

  // reference quirk: out.swapaxes(1,2).reshape(b,NUM,N,C):
  // t=2h+m; m'=t/8; n'=(t%8)*512 + n/8; c'=(n%8)*32 + d
  const int t2 = 2 * h + m;
  const int mp = t2 >> 3, tt = t2 & 7;
#pragma unroll
  for (int nn = 0; nn < 2; ++nn) {
#pragma unroll
    for (int r = 0; r < 8; ++r) {
      const int n = qt * 16 + r + rh;
      const int d = nn * 16 + rl;
      const float val = (nn == 0 ? o0[r] : o1[r]) * inv[r];
      const int np = tt * 512 + (n >> 3);
      const int cp = (n & 7) * 32 + d;
      aout[(size_t)b * 2097152 + (size_t)mp * 1048576 + (size_t)np * 256 + cp] = (bf16)val;
    }
  }
}

// ---------------------------------------------------------------------------
// Host launcher
// ---------------------------------------------------------------------------
extern "C" void kernel_launch(void* const* d_in, const int* in_sizes, int n_in,
                              void* d_out, int out_size, void* d_ws, size_t ws_size,
                              hipStream_t stream) {
  const float* x      = (const float*)d_in[0];   // (4,2,4096,256)
  const float* q_w    = (const float*)d_in[1];   // (256,256)
  const float* kv_w   = (const float*)d_in[2];   // (512,256)
  const float* proj_w = (const float*)d_in[3];   // (256,256)
  const float* proj_b = (const float*)d_in[4];   // (256)
  const float* sr_w   = (const float*)d_in[5];   // (256,1,4,4)
  const float* sr_b   = (const float*)d_in[6];   // (256)
  const float* ln_g   = (const float*)d_in[7];   // (256)
  const float* ln_b   = (const float*)d_in[8];   // (256)
  float* out = (float*)d_out;

  // Workspace carve-up (bf16 scratch, ~51 MB total).
  char* p = (char*)d_ws;
  bf16* x_bf    = (bf16*)p; p += (size_t)8388608 * 2;   // x as bf16
  bf16* q_bf    = (bf16*)p; p += (size_t)8388608 * 2;   // q = x@q_w^T * scale
  bf16* attn_bf = (bf16*)p; p += (size_t)8388608 * 2;   // attention out (scrambled layout)
  bf16* xr_bf   = (bf16*)p; p += (size_t)524288  * 2;   // conv+LN (2048 x 256)
  bf16* k_bf    = (bf16*)p; p += (size_t)524288  * 2;   // K per head (n_kv,32)
  bf16* vT_bf   = (bf16*)p; p += (size_t)524288  * 2;   // V^T per head (32,n_kv)
  bf16* qw_bf   = (bf16*)p; p += (size_t)65536   * 2;
  bf16* kvw_bf  = (bf16*)p; p += (size_t)131072  * 2;
  bf16* pw_bf   = (bf16*)p; p += (size_t)65536   * 2;

  const float scale = 0.17677669529663688f;  // 32^-0.5, folded into q_w

  cvt_bf16<<<2048, 256, 0, stream>>>(x, x_bf, 8388608, 1.f);
  cvt_bf16<<<64,   256, 0, stream>>>(q_w, qw_bf, 65536, scale);
  cvt_bf16<<<128,  256, 0, stream>>>(kv_w, kvw_bf, 131072, 1.f);
  cvt_bf16<<<64,   256, 0, stream>>>(proj_w, pw_bf, 65536, 1.f);

  conv_ln_kernel<<<2048, 256, 0, stream>>>(x, sr_w, sr_b, ln_g, ln_b, xr_bf);

  // kv: M=2048, N=512
  gemm_bf16_kernel<1><<<dim3(32, 8), 128, 0, stream>>>(
      xr_bf, kvw_bf, nullptr, k_bf, vT_bf, nullptr, nullptr);

  // q: M=32768, N=256
  gemm_bf16_kernel<0><<<dim3(512, 4), 128, 0, stream>>>(
      x_bf, qw_bf, q_bf, nullptr, nullptr, nullptr, nullptr);

  // attention: 64 (bm,h) x 256 q-tiles, one wave each
  attn_kernel<<<dim3(64, 256), 32, 0, stream>>>(q_bf, k_bf, vT_bf, attn_bf);

  // out projection: M=32768, N=256, +bias, f32 out
  gemm_bf16_kernel<2><<<dim3(512, 4), 128, 0, stream>>>(
      attn_bf, pw_bf, nullptr, nullptr, nullptr, proj_b, out);
}
